// Lenet5_85315230367862
// MI455X (gfx1250) — matrix operand order repaired and checked by reading, not compiled, not run on hardware
//
#include <hip/hip_runtime.h>
#include <hip/hip_bf16.h>

typedef __attribute__((ext_vector_type(16))) _Float16 v16h;
typedef __attribute__((ext_vector_type(8)))  float    v8f;

#define WMMA_F16(a, b, c) \
    __builtin_amdgcn_wmma_f32_16x16x32_f16(false, (a), false, (b), (short)0, (c), false, false)

// fast sigmoid: v_exp_f32 + v_rcp_f32 (precision already bounded by f16 WMMA inputs)
__device__ __forceinline__ float sigmoidf_(float x) {
    return __builtin_amdgcn_rcpf(1.0f + __expf(-x));
}

// ---------------- workspace layout (units: _Float16 elements) ----------------
// B-operand layouts: [n_tile][k_chunk] blocks of 512 halves; within a block,
// index = n_in_tile*32 + k_in_chunk  (lane reads 16 contiguous K halves).
static constexpr int B1_OFF  = 0;                    // 512    (K=32,  N=16;  6x25 valid)
static constexpr int B2_OFF  = 512;                  // 2560   (K=160, N=16;  16x150 valid)
static constexpr int B3_OFF  = 3072;                 // 53248  (K=416, N=128; 120x400 valid)
static constexpr int FC1_OFF = 56320;                // 12288  (K=128, N=96;  84x120 valid)
static constexpr int FC2_OFF = 68608;                // 1536   (K=96,  N=16;  10x84 valid)
static constexpr int ACT_OFF = 70144;                // N*416  pool2 activations (padded)
static constexpr long long ACT_STRIDE  = 416;
static constexpr long long H120_STRIDE = 128;
static constexpr long long H84_STRIDE  = 96;

// ---------------- kernel 1: effective weights -> WMMA B layouts ----------------
__global__ void prep_weights(const float* __restrict__ w1, const float* __restrict__ sg1, const int* __restrict__ sr1,
                             const float* __restrict__ w2, const float* __restrict__ sg2, const int* __restrict__ sr2,
                             const float* __restrict__ w3, const float* __restrict__ sg3, const int* __restrict__ sr3,
                             const float* __restrict__ fc1w, const float* __restrict__ fc2w,
                             _Float16* __restrict__ ws)
{
    int idx = blockIdx.x * blockDim.x + threadIdx.x;
    const int T1 = 512, T2 = T1 + 2560, T3 = T2 + 53248, T4 = T3 + 12288, T5 = T4 + 1536;
    if (idx >= T5) return;
    float val = 0.0f;
    int dst;
    if (idx < T1) {                                   // conv1 B: 6 x 25
        int n = idx >> 5, k = idx & 31;
        if (n < 6 && k < 25) { int f = n * 25 + k; val = sg1[f] * w1[n * 25 + sr1[f]]; }
        dst = B1_OFF + idx;
    } else if (idx < T2) {                            // conv2 B: 16 x 150
        int i = idx - T1;
        int chunk = i / 512, r = i & 511, n = r >> 5, k = chunk * 32 + (r & 31);
        if (k < 150) { int f = n * 150 + k; val = sg2[f] * w2[n * 150 + sr2[f]]; }
        dst = B2_OFF + i;
    } else if (idx < T3) {                            // conv3 B: 120 x 400
        int i = idx - T2;
        int g = i / 512, nt = g / 13, c = g % 13, r = i & 511;
        int n = nt * 16 + (r >> 5), k = c * 32 + (r & 31);
        if (n < 120 && k < 400) { int f = n * 400 + k; val = sg3[f] * w3[n * 400 + sr3[f]]; }
        dst = B3_OFF + i;
    } else if (idx < T4) {                            // fc1 B: 84 x 120
        int i = idx - T3;
        int g = i / 512, nt = g >> 2, c = g & 3, r = i & 511;
        int n = nt * 16 + (r >> 5), k = c * 32 + (r & 31);
        if (n < 84 && k < 120) val = fc1w[n * 120 + k];
        dst = FC1_OFF + i;
    } else {                                          // fc2 B: 10 x 84
        int i = idx - T4;
        int c = i / 512, r = i & 511;
        int n = r >> 5, k = c * 32 + (r & 31);
        if (n < 10 && k < 84) val = fc2w[n * 84 + k];
        dst = FC2_OFF + i;
    }
    ws[dst] = (_Float16)val;
}

// ---------------- kernel 2: fused conv1+pool1+conv2+pool2 (one block / image) ----------------
__global__ __launch_bounds__(128) void conv12_kernel(const float* __restrict__ x,
                                                     const float* __restrict__ b1,
                                                     const float* __restrict__ b2,
                                                     const _Float16* __restrict__ ws,
                                                     _Float16* __restrict__ act)
{
    __shared__ _Float16 xs[1024];         // 32x32 input, f16
    __shared__ _Float16 c1s[16 * 784];    // sigmoid(conv1), padded to 16 rows (branchless store)
    __shared__ _Float16 p1s[6 * 196];     // pool1 6x14x14
    __shared__ _Float16 c2s[16 * 112];    // sigmoid(conv2), padded to 112 pixels

    const int tid  = threadIdx.x;
    const int lane = tid & 31;
    const int wid  = tid >> 5;
    const int img  = blockIdx.x;
    const int mrow  = lane & 15;
    const int kbase = (lane < 16) ? 0 : 16;     // A-operand K: contiguous 16 per lane half
    const int sel   = (lane < 16) ? 0 : 16;     // B-operand K half-select
    const int mofs  = (lane < 16) ? 0 : 8;      // C-operand row offset

    // B operands + biases in registers, hoisted out of all loops
    v16h B1v = *(const v16h*)(ws + B1_OFF + mrow * 32 + sel);
    v16h B2v[5];
#pragma unroll
    for (int c = 0; c < 5; ++c)
        B2v[c] = *(const v16h*)(ws + B2_OFF + c * 512 + mrow * 32 + sel);
    const float bias1 = b1[mrow < 6 ? mrow : 0];
    const float bias2 = b2[mrow];

    // per-lane constant conv1 gather table: K -> (dy*32+dx), validity
    int  off1[16];
    bool ok1[16];
#pragma unroll
    for (int j = 0; j < 16; ++j) {
        int k = kbase + j;
        ok1[j]  = (k < 25);
        int kk  = ok1[j] ? k : 0;
        off1[j] = (kk / 5) * 32 + (kk % 5);
    }

    for (int i = tid; i < 1024; i += 128) xs[i] = (_Float16)x[(size_t)img * 1024 + i];
    __syncthreads();

    // ---- conv1: M=784 (49 tiles), K=25->32, N=6->16 ----
    for (int t = wid; t < 49; t += 4) {
        int p = t * 16 + mrow, r = p / 28, cc = p % 28;
        int base = r * 32 + cc;
        v16h a;
#pragma unroll
        for (int j = 0; j < 16; ++j) {
            _Float16 v = xs[base + off1[j]];          // unconditional ds_load
            a[j] = ok1[j] ? v : (_Float16)0.0f;       // v_cndmask
        }
        v8f c = {};
        c = WMMA_F16(a, B1v, c);
#pragma unroll
        for (int rr = 0; rr < 8; ++rr)                // unconditional store (rows 6..15 = scratch)
            c1s[mrow * 784 + t * 16 + rr + mofs] = (_Float16)sigmoidf_(c[rr] + bias1);
    }
    __syncthreads();

    // ---- pool1: 6x28x28 -> 6x14x14 ----
    for (int o = tid; o < 1176; o += 128) {
        int ch = o / 196, rem = o % 196, i = rem / 14, j = rem % 14;
        const _Float16* q = &c1s[ch * 784 + (2 * i) * 28 + 2 * j];
        p1s[o] = (_Float16)(0.25f * ((float)q[0] + (float)q[1] + (float)q[28] + (float)q[29]));
    }
    __syncthreads();

    // ---- conv2: M=100 (7 tiles), K=150->5x32, N=16 ----
    for (int t = wid; t < 7; t += 4) {
        int p = t * 16 + mrow;
        bool pv = (p < 100);
        int pp2 = pv ? p : 0;
        int pr = pp2 / 10, pc = pp2 % 10;
        int pbase = pr * 14 + pc;
        v8f c = {};
#pragma unroll
        for (int ck = 0; ck < 5; ++ck) {
            v16h a;
#pragma unroll
            for (int j = 0; j < 16; ++j) {
                int k  = ck * 32 + kbase + j;
                bool ok = pv && (k < 150);
                int kk  = (k < 150) ? k : 0;
                int ch  = kk / 25, kr = (kk % 25) / 5, kc = kk % 5;
                _Float16 v = p1s[ch * 196 + pbase + kr * 14 + kc];  // unconditional ds_load
                a[j] = ok ? v : (_Float16)0.0f;
            }
            c = WMMA_F16(a, B2v[ck], c);
        }
#pragma unroll
        for (int rr = 0; rr < 8; ++rr)                // unconditional store (pixels 100..111 = pad)
            c2s[mrow * 112 + t * 16 + rr + mofs] = (_Float16)sigmoidf_(c[rr] + bias2);
    }
    __syncthreads();

    // ---- pool2: 16x10x10 -> 16x5x5, write padded 416-wide activation row ----
    _Float16* arow = act + (size_t)img * ACT_STRIDE;
    for (int o = tid; o < 416; o += 128) {
        int oo = (o < 400) ? o : 0;
        int ch = oo / 25, rem = oo % 25, i = rem / 5, j = rem % 5;
        const _Float16* q = &c2s[ch * 112 + (2 * i) * 10 + 2 * j];
        float s = 0.25f * ((float)q[0] + (float)q[1] + (float)q[10] + (float)q[11]);
        arow[o] = (o < 400) ? (_Float16)s : (_Float16)0.0f;
    }
}

// ---------------- kernel 3: conv3-as-FC  sigmoid(act[N,400] @ W3^T + b3) ----------------
__global__ __launch_bounds__(128) void gemm_conv3(const _Float16* __restrict__ act,
                                                  const _Float16* __restrict__ b3B,
                                                  const float* __restrict__ b3,
                                                  _Float16* __restrict__ h120)
{
    const int lane = threadIdx.x & 31;
    const int wid  = threadIdx.x >> 5;
    const int img0 = (blockIdx.x * 4 + wid) * 16;
    const int col  = lane & 15;
    const int sel  = (lane < 16) ? 0 : 16;
    const int mofs = (lane < 16) ? 0 : 8;

    const _Float16* arow = act + (size_t)(img0 + col) * ACT_STRIDE + sel;
    v8f acc[8] = {};
    for (int c = 0; c < 13; ++c) {
        v16h a = *(const v16h*)(arow + c * 32);
        if (c + 1 < 13) __builtin_prefetch((const void*)(arow + (c + 1) * 32), 0, 0);
#pragma unroll
        for (int nt = 0; nt < 8; ++nt) {
            v16h b = *(const v16h*)(b3B + (nt * 13 + c) * 512 + col * 32 + sel);
            acc[nt] = WMMA_F16(a, b, acc[nt]);
        }
    }
#pragma unroll
    for (int nt = 0; nt < 8; ++nt) {
        int n = nt * 16 + col;
        float bias = b3[n < 120 ? n : 0];             // clamped, unconditional load
#pragma unroll
        for (int rr = 0; rr < 8; ++rr) {
            size_t img = (size_t)(img0 + rr + mofs);
            float s = sigmoidf_(acc[nt][rr] + bias);
            h120[img * H120_STRIDE + n] = (n < 120) ? (_Float16)s : (_Float16)0.0f;
        }
    }
}

// ---------------- kernel 4: fc1  sigmoid(h120 @ fc1_w^T + fc1_b) ----------------
__global__ __launch_bounds__(128) void gemm_fc1(const _Float16* __restrict__ h120,
                                                const _Float16* __restrict__ fc1B,
                                                const float* __restrict__ fc1b,
                                                _Float16* __restrict__ h84)
{
    const int lane = threadIdx.x & 31;
    const int wid  = threadIdx.x >> 5;
    const int img0 = (blockIdx.x * 4 + wid) * 16;
    const int col  = lane & 15;
    const int sel  = (lane < 16) ? 0 : 16;
    const int mofs = (lane < 16) ? 0 : 8;

    const _Float16* arow = h120 + (size_t)(img0 + col) * H120_STRIDE + sel;
    v8f acc[6] = {};
#pragma unroll
    for (int c = 0; c < 4; ++c) {
        v16h a = *(const v16h*)(arow + c * 32);
#pragma unroll
        for (int nt = 0; nt < 6; ++nt) {
            v16h b = *(const v16h*)(fc1B + (nt * 4 + c) * 512 + col * 32 + sel);
            acc[nt] = WMMA_F16(a, b, acc[nt]);
        }
    }
#pragma unroll
    for (int nt = 0; nt < 6; ++nt) {
        int n = nt * 16 + col;
        float bias = fc1b[n < 84 ? n : 0];
#pragma unroll
        for (int rr = 0; rr < 8; ++rr) {
            size_t img = (size_t)(img0 + rr + mofs);
            float s = sigmoidf_(acc[nt][rr] + bias);
            h84[img * H84_STRIDE + n] = (n < 84) ? (_Float16)s : (_Float16)0.0f;
        }
    }
}

// ---------------- kernel 5: fc2  h84 @ fc2_w^T + fc2_b  -> d_out (f32) ----------------
__global__ __launch_bounds__(128) void gemm_fc2(const _Float16* __restrict__ h84,
                                                const _Float16* __restrict__ fc2B,
                                                const float* __restrict__ fc2b,
                                                float* __restrict__ out)
{
    const int lane = threadIdx.x & 31;
    const int wid  = threadIdx.x >> 5;
    const int img0 = (blockIdx.x * 4 + wid) * 16;
    const int col  = lane & 15;
    const int sel  = (lane < 16) ? 0 : 16;
    const int mofs = (lane < 16) ? 0 : 8;

    const _Float16* arow = h84 + (size_t)(img0 + col) * H84_STRIDE + sel;
    const float bias = fc2b[col < 10 ? col : 0];
    v8f acc = {};
#pragma unroll
    for (int c = 0; c < 3; ++c) {
        v16h a = *(const v16h*)(arow + c * 32);
        v16h b = *(const v16h*)(fc2B + c * 512 + col * 32 + sel);
        acc = WMMA_F16(a, b, acc);
    }
    if (col < 10) {                                   // single guarded region (d_out stores)
#pragma unroll
        for (int rr = 0; rr < 8; ++rr) {
            size_t img = (size_t)(img0 + rr + mofs);
            out[img * 10 + col] = acc[rr] + bias;
        }
    }
}

extern "C" void kernel_launch(void* const* d_in, const int* in_sizes, int n_in,
                              void* d_out, int out_size, void* d_ws, size_t ws_size,
                              hipStream_t stream) {
    const float* x     = (const float*)d_in[0];
    const float* w1    = (const float*)d_in[1];
    const float* b1    = (const float*)d_in[2];
    const float* sg1   = (const float*)d_in[3];
    const float* w2    = (const float*)d_in[4];
    const float* b2    = (const float*)d_in[5];
    const float* sg2   = (const float*)d_in[6];
    const float* w3    = (const float*)d_in[7];
    const float* b3    = (const float*)d_in[8];
    const float* sg3   = (const float*)d_in[9];
    const float* fc1w  = (const float*)d_in[10];
    const float* fc1b  = (const float*)d_in[11];
    const float* fc2w  = (const float*)d_in[12];
    const float* fc2b  = (const float*)d_in[13];
    const int*   sr1   = (const int*)d_in[14];
    const int*   sr2   = (const int*)d_in[15];
    const int*   sr3   = (const int*)d_in[16];
    // TRIGGERed==1 in setup_inputs(); effective-weight path implemented.

    const int N = in_sizes[0] / 1024;           // 16384 images
    _Float16* ws = (_Float16*)d_ws;
    _Float16* act  = ws + ACT_OFF;
    _Float16* h120 = act + (long long)N * ACT_STRIDE;
    _Float16* h84  = h120 + (long long)N * H120_STRIDE;

    const int prep_total = 512 + 2560 + 53248 + 12288 + 1536;
    prep_weights<<<(prep_total + 255) / 256, 256, 0, stream>>>(
        w1, sg1, sr1, w2, sg2, sr2, w3, sg3, sr3, fc1w, fc2w, ws);

    conv12_kernel<<<N, 128, 0, stream>>>(x, b1, b2, ws, act);

    const int mblocks = N / 64;                 // 4 waves/block, 16 images/wave
    gemm_conv3<<<mblocks, 128, 0, stream>>>(act, ws + B3_OFF, b3, h120);
    gemm_fc1 <<<mblocks, 128, 0, stream>>>(h120, ws + FC1_OFF, fc1b, h84);
    gemm_fc2 <<<mblocks, 128, 0, stream>>>(h84, ws + FC2_OFF, fc2b, (float*)d_out);
}
